// DecoderOnly_11149735100530
// MI455X (gfx1250) — matrix operand order repaired and checked by reading, not compiled, not run on hardware
//
#include <hip/hip_runtime.h>
#include <hip/hip_bf16.h>
#include <math.h>

#define S_   24
#define B_   256
#define D_   1024
#define H_   16
#define L_   12
#define V_   32000
#define DH_  64
#define NROW (S_ * B_)   // 6144 rows in the flattened [S*B, D] activation matrix

#define BSTRIDE 40       // padded LDS row stride (bf16): 32 data + 8 pad = 80B

// Tensor Data Mover availability / arity probe
#if defined(__has_builtin)
#  if __has_builtin(__builtin_amdgcn_tensor_load_to_lds)
#    define USE_TDM 1
#  endif
#endif
#ifndef USE_TDM
#  define USE_TDM 0
#endif

typedef __bf16 bf16;
typedef __attribute__((ext_vector_type(16))) __bf16    v16bf;
typedef __attribute__((ext_vector_type(8)))  __bf16    v8bf;
typedef __attribute__((ext_vector_type(8)))  float     v8f;
typedef __attribute__((ext_vector_type(4)))  unsigned  v4u;
typedef __attribute__((ext_vector_type(8)))  int       v8i;
typedef __attribute__((ext_vector_type(4)))  int       v4i;

__device__ __forceinline__ bf16 f2bf(float f) {
    unsigned u = __builtin_bit_cast(unsigned, f);
    unsigned r = (u + 0x7FFFu + ((u >> 16) & 1u)) >> 16;   // round-to-nearest-even
    return __builtin_bit_cast(bf16, (unsigned short)r);
}
__device__ __forceinline__ float bf2f(bf16 b) {
    unsigned u = ((unsigned)__builtin_bit_cast(unsigned short, b)) << 16;
    return __builtin_bit_cast(float, u);
}

#if USE_TDM
// Issue TENSOR_LOAD_TO_LDS of a 64(N-rows) x 32(K) bf16 tile starting at gsrc
// (row stride K elems) into LDS offset lds_off, with HW padding of 4 DWORDs
// after every 16 DWORDs -> padded LDS row stride of 80B (= BSTRIDE bf16).
// D# layout per CDNA5 ISA 8.3/8.4 (group0 128b, group1 256b; 2D => g2/g3 zero).
__device__ __forceinline__ void tdm_load_b_tile(const bf16* gsrc, unsigned lds_off, int K) {
    unsigned long long ga = (unsigned long long)(size_t)gsrc;
    v4u g0;
    g0.x = 1u;                                   // count=1, is_restore=0, no gather
    g0.y = lds_off;                              // lds_addr [63:32]
    g0.z = (unsigned)ga;                         // global_addr [95:64]
    g0.w = (unsigned)((ga >> 32) & 0x1FFFFFFu)   // global_addr [120:96]
         | (2u << 30);                           // type=2 ("image")
    const unsigned td0 = (unsigned)K;            // tensor_dim0 (elems / row)
    const unsigned td1 = 64u;                    // tensor_dim1 (rows)
    v8i g1;
    g1.s0 = (1 << 16)                            // data_size = 2 bytes
          | (1 << 20)                            // pad_enable
          | (3 << 22)                            // pad_interval: 16 DWORDs
          | (3 << 25);                           // pad_amount: 4 DWORDs
    g1.s1 = (int)((td0 & 0xFFFFu) << 16);        // tensor_dim0[15:0]
    g1.s2 = (int)(((td0 >> 16) & 0xFFFFu) | ((td1 & 0xFFFFu) << 16));
    g1.s3 = (int)(((td1 >> 16) & 0xFFFFu) | (32u << 16));   // tile_dim0 = 32
    g1.s4 = (int)64;                             // tile_dim1 = 64, tile_dim2 = 0
    g1.s5 = (int)td0;                            // tensor_dim0_stride[31:0] = K
    g1.s6 = 0;                                   // stride0[47:32]=0, stride1 lo=0
    g1.s7 = 0;                                   // stride1 hi = 0
    v4i z4 = {0, 0, 0, 0};
#if defined(__clang_major__) && __clang_major__ >= 23
    v8i z8 = {0, 0, 0, 0, 0, 0, 0, 0};
    __builtin_amdgcn_tensor_load_to_lds(g0, g1, z4, z4, z8, 0);
#else
    __builtin_amdgcn_tensor_load_to_lds(g0, g1, z4, z4, 0);
#endif
}
#endif

// ---------------------------------------------------------------------------
// fp32 -> bf16 bulk conversion (weights, once per call)
// ---------------------------------------------------------------------------
__global__ __launch_bounds__(256) void cvt_f32_bf16(const float* __restrict__ src,
                                                    bf16* __restrict__ dst, long n) {
    long i = (long)blockIdx.x * blockDim.x + threadIdx.x;
    if (i < n) dst[i] = f2bf(src[i]);
}

// ---------------------------------------------------------------------------
// Embedding gather + positional encoding -> x (fp32 residual) and xb (bf16)
// ---------------------------------------------------------------------------
__global__ __launch_bounds__(256) void embed_kernel(const int* __restrict__ tok,
                                                    const float* __restrict__ te,
                                                    const float* __restrict__ pe,
                                                    float* __restrict__ x,
                                                    bf16* __restrict__ xb) {
    long i = (long)blockIdx.x * blockDim.x + threadIdx.x;   // flat over S*B*D
    if (i >= (long)NROW * D_) return;
    int d  = (int)(i % D_);
    long r = i / D_;            // r = s*B + b
    int s  = (int)(r / B_);
    int t  = tok[r];
    float v = te[(long)t * D_ + d] + pe[(long)s * D_ + d];
    x[i]  = v;
    xb[i] = f2bf(v);
}

// ---------------------------------------------------------------------------
// bf16 WMMA GEMM:  C[M,N] = A[M,K] @ Bw[N,K]^T + bias[N]
//
// Block = 256 threads (8 waves). Block tile = 256(M) x 64(N).
// Wave tile = 32(M) x 64(N): 2 A fragments x 4 B fragments = 8 WMMAs / K-step.
// The 32(K)x64(N) B tile (shared by all 8 waves) is double-buffered in LDS.
// With TDM: wave 0 issues one tensor_load_to_lds per K-step for the next
// buffer (HW-padded rows), waits TENSORcnt, and the workgroup barrier
// publishes it; the vector pipe only issues A loads + ds reads + WMMAs.
// Dims must satisfy M%256==0, N%64==0, K%32==0 (true for all calls here).
// ---------------------------------------------------------------------------
template <typename OutT>
__global__ __launch_bounds__(256) void gemm_bf16_wmma(const bf16* __restrict__ A,
                                                      const bf16* __restrict__ Bw,
                                                      const float* __restrict__ bias,
                                                      OutT* __restrict__ C,
                                                      int M, int N, int K) {
    const int tid  = threadIdx.x;
    const int lane = tid & 31;
    const int wave = tid >> 5;
    const int half = lane >> 4;     // 0: lanes 0-15, 1: lanes 16-31
    const int r    = lane & 15;

    const int m0 = blockIdx.y * 256 + wave * 32;   // wave's M origin
    const int n0 = blockIdx.x * 64;                // block's N origin

    // LDS double buffer for the B tile: 64 N-rows x 32 K (padded stride)
    __shared__ bf16 lB[2][64 * BSTRIDE];

    v8f acc[2][4] = {};

    const bf16* aRow0 = A + (size_t)(m0 + r) * K;        // M rows r
    const bf16* aRow1 = A + (size_t)(m0 + 16 + r) * K;   // M rows 16+r
    const bf16* bTile = Bw + (size_t)n0 * K;             // B tile row 0

#if USE_TDM
    const unsigned ldsOff[2] = { (unsigned)(size_t)&lB[0][0],
                                 (unsigned)(size_t)&lB[1][0] };
    if (wave == 0) tdm_load_b_tile(bTile, ldsOff[0], K);   // prologue: buf 0
#else
    // Cooperative fallback: each thread owns one 16B chunk of the tile.
    const int brow = tid >> 2;          // 0..63  (N-row within tile)
    const int bcol = (tid & 3) * 8;     // 0,8,16,24 (K offset within step)
    const bf16* bSrc = bTile + (size_t)brow * K + bcol;
    {
        v8bf b0 = *(const v8bf*)(bSrc);
        *(v8bf*)(&lB[0][brow * BSTRIDE + bcol]) = b0;
    }
#endif

    int cur = 0;
    for (int k0 = 0; k0 < K; k0 += 32) {
        const bool more = (k0 + 32) < K;

#if USE_TDM
        if (wave == 0) __builtin_amdgcn_s_wait_tensorcnt(0);  // buf `cur` landed
        __syncthreads();                                      // publish to all
        // Prior reads of buf cur^1 finished before this barrier -> safe to DMA.
        if (more && wave == 0) tdm_load_b_tile(bTile + k0 + 32, ldsOff[cur ^ 1], K);
#else
        __syncthreads();   // buffer `cur` published; prev iter's reads done
#endif

        // A fragments FIRST (WMMAs wait only on these + ds, while any
        // in-flight next-tile traffic continues behind them).
        v8bf a00 = *(const v8bf*)(aRow0 + k0 + half * 8);
        v8bf a01 = *(const v8bf*)(aRow0 + k0 + half * 8 + 16);
        v8bf a10 = *(const v8bf*)(aRow1 + k0 + half * 8);
        v8bf a11 = *(const v8bf*)(aRow1 + k0 + half * 8 + 16);
        v16bf af0, af1;
#pragma unroll
        for (int i = 0; i < 8; ++i) {
            af0[i] = a00[i]; af0[i + 8] = a01[i];
            af1[i] = a10[i]; af1[i + 8] = a11[i];
        }

#if !USE_TDM
        // Next B tile -> registers, issued AFTER the A loads (in-order
        // completion: waiting for A no longer drains this load too).
        v8bf bnext = *(const v8bf*)(bSrc + (more ? (k0 + 32) : 0));
#endif

        // B fragments from LDS: lane holds column f*16+r, K = half*8..+7 and 16+half*8..+7
        v16bf bfrag[4];
#pragma unroll
        for (int f = 0; f < 4; ++f) {
            const bf16* lp = &lB[cur][(f * 16 + r) * BSTRIDE + half * 8];
            v8bf b0 = *(const v8bf*)(lp);
            v8bf b1 = *(const v8bf*)(lp + 16);
#pragma unroll
            for (int i = 0; i < 8; ++i) { bfrag[f][i] = b0[i]; bfrag[f][i + 8] = b1[i]; }
        }

#pragma unroll
        for (int f = 0; f < 4; ++f) {
            acc[0][f] = __builtin_amdgcn_wmma_f32_16x16x32_bf16(
                false, af0, false, bfrag[f], (short)0, acc[0][f], false, false);
            acc[1][f] = __builtin_amdgcn_wmma_f32_16x16x32_bf16(
                false, af1, false, bfrag[f], (short)0, acc[1][f], false, false);
        }

#if !USE_TDM
        if (more) *(v8bf*)(&lB[cur ^ 1][brow * BSTRIDE + bcol]) = bnext;
#endif
        if (more) cur ^= 1;
    }

    // Epilogue. C/D layout: VGPR i -> (M = half*8 + i, N = lane%16)
#pragma unroll
    for (int g = 0; g < 2; ++g) {
#pragma unroll
        for (int f = 0; f < 4; ++f) {
            const int col = n0 + f * 16 + r;
            const float bv = bias[col];
#pragma unroll
            for (int i = 0; i < 8; ++i) {
                const int row = m0 + g * 16 + half * 8 + i;
                float val = acc[g][f][i] + bv;
                if constexpr (__is_same(OutT, float)) {
                    C[(size_t)row * N + col] = val;
                } else {
                    C[(size_t)row * N + col] = f2bf(val);
                }
            }
        }
    }
}

// ---------------------------------------------------------------------------
// Attention for one (b,h): scores = qk^T/sqrt(D); softmax; o = attn@v;
// fused residual x += o and bf16 refresh of xb. One wave per (b,h).
// ---------------------------------------------------------------------------
__global__ __launch_bounds__(32) void attn_kernel(const bf16* __restrict__ q,
                                                  const bf16* __restrict__ k,
                                                  const bf16* __restrict__ v,
                                                  float* __restrict__ x,
                                                  bf16* __restrict__ xb) {
    const int bh = blockIdx.x;
    const int b  = bh / H_;
    const int h  = bh % H_;
    const int tid = threadIdx.x;

    __shared__ bf16 lq[S_ * DH_];
    __shared__ bf16 lk[S_ * DH_];
    __shared__ bf16 lv[S_ * DH_];

    for (int i = tid; i < S_ * DH_; i += 32) {
        int s  = i >> 6;
        int dh = i & 63;
        size_t g = ((size_t)(s * B_ + b)) * D_ + h * DH_ + dh;
        lq[i] = q[g];
        lk[i] = k[g];
        lv[i] = v[g];
    }
    __syncthreads();

    if (tid < S_) {
        const int s = tid;
        const float scale = 0.03125f;   // 1/sqrt(1024)
        float sc[S_];
        float mx = -3.4e38f;
        for (int j = 0; j < S_; ++j) {
            float d = 0.f;
#pragma unroll
            for (int t = 0; t < DH_; ++t)
                d += bf2f(lq[s * DH_ + t]) * bf2f(lk[j * DH_ + t]);
            d *= scale;
            sc[j] = d;
            mx = fmaxf(mx, d);
        }
        float sum = 0.f;
        for (int j = 0; j < S_; ++j) { sc[j] = __expf(sc[j] - mx); sum += sc[j]; }
        const float inv = 1.0f / sum;

        float o[DH_];
#pragma unroll
        for (int t = 0; t < DH_; ++t) o[t] = 0.f;
        for (int j = 0; j < S_; ++j) {
            const float w = sc[j] * inv;
#pragma unroll
            for (int t = 0; t < DH_; ++t) o[t] += w * bf2f(lv[j * DH_ + t]);
        }

        const size_t base = ((size_t)(s * B_ + b)) * D_ + h * DH_;
#pragma unroll
        for (int t = 0; t < DH_; ++t) {
            float nx = x[base + t] + o[t];
            x[base + t]  = nx;
            xb[base + t] = f2bf(nx);
        }
    }
}

// ---------------------------------------------------------------------------
// Host launcher
// ---------------------------------------------------------------------------
extern "C" void kernel_launch(void* const* d_in, const int* in_sizes, int n_in,
                              void* d_out, int out_size, void* d_ws, size_t ws_size,
                              hipStream_t stream) {
    (void)in_sizes; (void)n_in; (void)out_size; (void)ws_size;

    const int*   tok   = (const int*)  d_in[0];
    // d_in[1] = mask (unused, matching reference which drops it)
    const float* te    = (const float*)d_in[2];
    const float* pe    = (const float*)d_in[3];
    const float* qw    = (const float*)d_in[4];
    const float* qb    = (const float*)d_in[5];
    const float* kw    = (const float*)d_in[6];
    const float* kb    = (const float*)d_in[7];
    const float* vw    = (const float*)d_in[8];
    const float* vb    = (const float*)d_in[9];
    const float* ow    = (const float*)d_in[10];
    const float* ob    = (const float*)d_in[11];
    float* logits = (float*)d_out;

    // Workspace layout (all sizes multiples of 256B)
    char* w = (char*)d_ws;
    float* x   = (float*)w;  w += (size_t)NROW * D_ * sizeof(float);
    bf16* xb   = (bf16*)w;   w += (size_t)NROW * D_ * sizeof(bf16);
    bf16* qact = (bf16*)w;   w += (size_t)NROW * D_ * sizeof(bf16);
    bf16* kact = (bf16*)w;   w += (size_t)NROW * D_ * sizeof(bf16);
    bf16* vact = (bf16*)w;   w += (size_t)NROW * D_ * sizeof(bf16);
    bf16* wqb  = (bf16*)w;   w += (size_t)L_ * D_ * D_ * sizeof(bf16);
    bf16* wkb  = (bf16*)w;   w += (size_t)L_ * D_ * D_ * sizeof(bf16);
    bf16* wvb  = (bf16*)w;   w += (size_t)L_ * D_ * D_ * sizeof(bf16);
    bf16* owb  = (bf16*)w;   w += (size_t)V_ * D_ * sizeof(bf16);

    // 1) Convert weights fp32 -> bf16 (once per call; ~280MB pass, ~12us at HBM rate)
    {
        long nw = (long)L_ * D_ * D_;
        int  blk = 256;
        cvt_f32_bf16<<<(unsigned)((nw + blk - 1) / blk), blk, 0, stream>>>(qw, wqb, nw);
        cvt_f32_bf16<<<(unsigned)((nw + blk - 1) / blk), blk, 0, stream>>>(kw, wkb, nw);
        cvt_f32_bf16<<<(unsigned)((nw + blk - 1) / blk), blk, 0, stream>>>(vw, wvb, nw);
        long no = (long)V_ * D_;
        cvt_f32_bf16<<<(unsigned)((no + blk - 1) / blk), blk, 0, stream>>>(ow, owb, no);
    }

    // 2) Embedding + PE
    {
        long n = (long)NROW * D_;
        embed_kernel<<<(unsigned)((n + 255) / 256), 256, 0, stream>>>(tok, te, pe, x, xb);
    }

    // 3) Decoder layers
    dim3 gQKV(D_ / 64, NROW / 256);   // (16, 24)
    for (int l = 0; l < L_; ++l) {
        const bf16* Wq = wqb + (size_t)l * D_ * D_;
        const bf16* Wk = wkb + (size_t)l * D_ * D_;
        const bf16* Wv = wvb + (size_t)l * D_ * D_;
        const float* bq = qb + (size_t)l * D_;
        const float* bk = kb + (size_t)l * D_;
        const float* bv = vb + (size_t)l * D_;

        gemm_bf16_wmma<bf16><<<gQKV, 256, 0, stream>>>(xb, Wq, bq, qact, NROW, D_, D_);
        gemm_bf16_wmma<bf16><<<gQKV, 256, 0, stream>>>(xb, Wk, bk, kact, NROW, D_, D_);
        gemm_bf16_wmma<bf16><<<gQKV, 256, 0, stream>>>(xb, Wv, bv, vact, NROW, D_, D_);

        attn_kernel<<<B_ * H_, 32, 0, stream>>>(qact, kact, vact, x, xb);
    }

    // 4) Final vocab projection -> fp32 logits
    dim3 gOut(V_ / 64, NROW / 256);   // (500, 24)
    gemm_bf16_wmma<float><<<gOut, 256, 0, stream>>>(xb, owb, ob, logits, NROW, V_, D_);
}
